// GNNGuard_30039001268952
// MI455X (gfx1250) — matrix operand order repaired and checked by reading, not compiled, not run on hardware
//
#include <hip/hip_runtime.h>
#include <hip/hip_bf16.h>
#include <math.h>

// ---------------------------------------------------------------------------
// Types for CDNA5 WMMA (wave32, 16x16x32 bf16 -> f32 accumulate)
// ---------------------------------------------------------------------------
typedef __attribute__((ext_vector_type(16))) __bf16 v16bf;
typedef __attribute__((ext_vector_type(8)))  __bf16 v8bf;
typedef __attribute__((ext_vector_type(8)))  float  v8f;

// ---------------------------------------------------------------------------
// f32 -> bf16 conversion (row-major copy)
// ---------------------------------------------------------------------------
__global__ void cvt_bf16_kernel(const float* __restrict__ in, __bf16* __restrict__ out,
                                long long n) {
    long long i = blockIdx.x * (long long)blockDim.x + threadIdx.x;
    if (i < n) out[i] = (__bf16)in[i];
}

// Transpose + convert + pad: Wt[c][k] = W[k][c] for c < C, else 0.  Wt is [Cpad x K].
__global__ void transpose_w_bf16_kernel(const float* __restrict__ W, __bf16* __restrict__ Wt,
                                        int K, int C, int Cpad) {
    int i = blockIdx.x * blockDim.x + threadIdx.x;
    if (i >= Cpad * K) return;
    int c = i / K, k = i % K;
    Wt[i] = (__bf16)((c < C) ? W[(size_t)k * C + c] : 0.0f);
}

// ---------------------------------------------------------------------------
// WMMA GEMM: C[M x Npad] = A[M x K] * B[K x Npad], with B given transposed
// (Bt[Npad x K]).  bf16 inputs, f32 accumulate.
// One wave computes one 16-row M-tile x NT 16-col N-tiles (A-fragment reuse).
// K is a template constant -> fully unrolled chain of v_wmma.
// Fragment layouts per ISA 7.12.2:
//   A (16x32, 16-bit): lane=M row; a[0..7]=K[half*8 .. +7], a[8..15]=K[16+half*8 .. +7]
//   B (32x16, 16-bit): lane=N col; b[0..7]=K[half*8 .. +7], b[8..15]=K[16+half*8 .. +7]
// With Bt row-major in K, both operands load as two global_load_b128 per fragment.
// All branches are wave-uniform (EXEC all-1s for WMMA).
// ---------------------------------------------------------------------------
template<int K, int NT>
__global__ void gemm_bf16_wmma_kernel(const __bf16* __restrict__ A,
                                      const __bf16* __restrict__ Bt,
                                      float* __restrict__ Cout,
                                      int Npad) {
    const int lane = threadIdx.x;                 // 0..31
    const int l    = lane & 15;
    const int half = lane >> 4;                   // 0 or 1
    const int m0   = blockIdx.y * 16;
    const int n0   = (blockIdx.x * blockDim.y + threadIdx.y) * (NT * 16);

    v8f acc[NT];
#pragma unroll
    for (int t = 0; t < NT; ++t) acc[t] = (v8f){};

    const __bf16* Arow = A + (size_t)(m0 + l) * K;

#pragma unroll
    for (int kk = 0; kk < K; kk += 32) {
        // A fragment: two contiguous 8-element (16B) chunks
        v8bf alo = *(const v8bf*)(Arow + kk + half * 8);
        v8bf ahi = *(const v8bf*)(Arow + kk + 16 + half * 8);
        v16bf a = __builtin_shufflevector(alo, ahi,
            0, 1, 2, 3, 4, 5, 6, 7, 8, 9, 10, 11, 12, 13, 14, 15);
#pragma unroll
        for (int t = 0; t < NT; ++t) {
            const __bf16* Brow = Bt + (size_t)(n0 + t * 16 + l) * K + kk + half * 8;
            v8bf blo = *(const v8bf*)(Brow);
            v8bf bhi = *(const v8bf*)(Brow + 16);
            v16bf b = __builtin_shufflevector(blo, bhi,
                0, 1, 2, 3, 4, 5, 6, 7, 8, 9, 10, 11, 12, 13, 14, 15);
            acc[t] = __builtin_amdgcn_wmma_f32_16x16x32_bf16(
                /*neg_a=*/false, a, /*neg_b=*/false, b,
                /*c_mod=*/(short)0, acc[t], /*reuse_a=*/false, /*reuse_b=*/false);
        }
    }
    // D layout: VGPR r -> row m0 + r + 8*half, col n0 + t*16 + l
#pragma unroll
    for (int t = 0; t < NT; ++t)
#pragma unroll
        for (int r = 0; r < 8; ++r)
            Cout[(size_t)(m0 + r + 8 * half) * Npad + (n0 + t * 16 + l)] = acc[t][r];
}

// ---------------------------------------------------------------------------
// Per-node L2 norms: one wave per node (256 feats -> 8 per lane, shfl reduce)
// ---------------------------------------------------------------------------
__global__ void norms_kernel(const float* __restrict__ X, float* __restrict__ nrm,
                             int N, int F) {
    int wid  = (blockIdx.x * blockDim.x + threadIdx.x) >> 5;
    int lane = threadIdx.x & 31;
    if (wid >= N) return;
    const float* xr = X + (size_t)wid * F;
    float s = 0.f;
    for (int k = lane; k < F; k += 32) s += xr[k] * xr[k];
    for (int off = 16; off; off >>= 1) s += __shfl_xor(s, off, 32);
    if (lane == 0) nrm[wid] = fmaxf(sqrtf(s), 1e-12f);
}

// ---------------------------------------------------------------------------
// Per-edge cosine similarity + threshold + rowsum accumulation (wave per edge)
// ---------------------------------------------------------------------------
__global__ void edge_sim_kernel(const float* __restrict__ X,
                                const int* __restrict__ row, const int* __restrict__ col,
                                const float* __restrict__ nrm,
                                float* __restrict__ sim, float* __restrict__ rowsum,
                                int E, int F) {
    int wid  = (blockIdx.x * blockDim.x + threadIdx.x) >> 5;
    int lane = threadIdx.x & 31;
    if (wid >= E) return;
    int r = row[wid], c = col[wid];
    const float* xr = X + (size_t)r * F;
    const float* xc = X + (size_t)c * F;
    float s = 0.f;
    for (int k = lane; k < F; k += 32) s += xr[k] * xc[k];
    for (int off = 16; off; off >>= 1) s += __shfl_xor(s, off, 32);
    if (lane == 0) {
        float v = s / (nrm[r] * nrm[c]);
        if (v < 0.1f || r == c) v = 0.f;
        sim[wid] = v;
        if (v != 0.f) atomicAdd(&rowsum[r], v);
    }
}

// sim_n = sim / rowsum[row] (rowsum==0 -> 1); count nonzeros into deg[row]
__global__ void edge_norm_kernel(float* __restrict__ sim,
                                 const int* __restrict__ row,
                                 const float* __restrict__ rowsum,
                                 float* __restrict__ deg, int E) {
    int e = blockIdx.x * blockDim.x + threadIdx.x;
    if (e >= E) return;
    int r = row[e];
    float rs = rowsum[r];
    rs = (rs == 0.f) ? 1.f : rs;
    float sn = sim[e] / rs;
    sim[e] = sn;
    if (sn > 0.f) atomicAdd(&deg[r], 1.0f);
}

// w_s = exp(1/(deg+1)); initialize weighted degree with self-loop weight
__global__ void node_ws_kernel(const float* __restrict__ deg,
                               float* __restrict__ wsv, float* __restrict__ wdeg, int N) {
    int i = blockIdx.x * blockDim.x + threadIdx.x;
    if (i >= N) return;
    float w = __expf(1.0f / (deg[i] + 1.0f));
    wsv[i]  = w;
    wdeg[i] = w;
}

// w_e = exp(sim_n) for stored nonzeros; accumulate weighted degree over col
__global__ void edge_we_kernel(const float* __restrict__ sim,
                               const int* __restrict__ col,
                               float* __restrict__ we, float* __restrict__ wdeg, int E) {
    int e = blockIdx.x * blockDim.x + threadIdx.x;
    if (e >= E) return;
    float sn = sim[e];
    float w  = (sn > 0.f) ? __expf(sn) : 0.f;
    we[e] = w;
    if (w != 0.f) atomicAdd(&wdeg[col[e]], w);
}

__global__ void node_dinv_kernel(const float* __restrict__ wdeg,
                                 float* __restrict__ dinv, int N) {
    int i = blockIdx.x * blockDim.x + threadIdx.x;
    if (i >= N) return;
    dinv[i] = rsqrtf(wdeg[i]);
}

// out[node,f] = dinv^2 * w_s * Hlin[node,f] + bias[f]  (self-loop + bias)
__global__ void agg_init_kernel(const float* __restrict__ Hlin,
                                const float* __restrict__ dinv, const float* __restrict__ wsv,
                                const float* __restrict__ bias, float* __restrict__ outp,
                                int N, int F, int strideH, int strideO) {
    long long i = blockIdx.x * (long long)blockDim.x + threadIdx.x;
    if (i >= (long long)N * F) return;
    int node = (int)(i / F), f = (int)(i % F);
    float d = dinv[node];
    outp[(size_t)node * strideO + f] =
        d * d * wsv[node] * Hlin[(size_t)node * strideH + f] + bias[f];
}

// Scatter edge messages: out[col] += dinv[row]*w_e*dinv[col] * Hlin[row]
__global__ void agg_edges_kernel(const float* __restrict__ Hlin,
                                 const int* __restrict__ row, const int* __restrict__ col,
                                 const float* __restrict__ we, const float* __restrict__ dinv,
                                 float* __restrict__ outp,
                                 int E, int F, int strideH, int strideO) {
    int wid  = (blockIdx.x * blockDim.x + threadIdx.x) >> 5;
    int lane = threadIdx.x & 31;
    if (wid >= E) return;
    float w = we[wid];
    if (w == 0.f) return;
    int r = row[wid], c = col[wid];
    float coef = dinv[r] * w * dinv[c];
    const float* hr = Hlin + (size_t)r * strideH;
    float*       oc = outp + (size_t)c * strideO;
    for (int k = lane; k < F; k += 32) atomicAdd(&oc[k], coef * hr[k]);
}

__global__ void relu_kernel(float* __restrict__ h, long long n) {
    long long i = blockIdx.x * (long long)blockDim.x + threadIdx.x;
    if (i < n) h[i] = fmaxf(h[i], 0.0f);
}

// In-place row-wise log_softmax over C classes (one thread per node)
__global__ void logsoftmax_kernel(float* __restrict__ io, int N, int C) {
    int i = blockIdx.x * blockDim.x + threadIdx.x;
    if (i >= N) return;
    float* p = io + (size_t)i * C;
    float m = -INFINITY;
    for (int c = 0; c < C; ++c) m = fmaxf(m, p[c]);
    float s = 0.f;
    for (int c = 0; c < C; ++c) s += __expf(p[c] - m);
    float lse = m + logf(s);
    for (int c = 0; c < C; ++c) p[c] -= lse;
}

// ---------------------------------------------------------------------------
// Orchestration
// ---------------------------------------------------------------------------
extern "C" void kernel_launch(void* const* d_in, const int* in_sizes, int n_in,
                              void* d_out, int out_size, void* d_ws, size_t ws_size,
                              hipStream_t stream) {
    const float* x  = (const float*)d_in[0];
    const int*   ei = (const int*)  d_in[1];
    const float* W1 = (const float*)d_in[2];
    const float* b1 = (const float*)d_in[3];
    const float* W2 = (const float*)d_in[4];
    const float* b2 = (const float*)d_in[5];
    float* out = (float*)d_out;

    const int F    = 256;                  // NFEAT == NHID (compile-time K below)
    const int N    = in_sizes[0] / F;      // 65536
    const int E    = in_sizes[1] / 2;      // 1048576
    const int C    = in_sizes[5];          // NCLASS = 40
    const int Cpad = (C + 15) & ~15;       // 48

    const int* row = ei;
    const int* col = ei + E;

    // ---- workspace carve (256B aligned) ----
    uintptr_t p = (uintptr_t)d_ws;
    auto take = [&](size_t bytes) -> void* {
        uintptr_t r = p; p += (bytes + 255) & ~(size_t)255; return (void*)r;
    };
    __bf16* xb   = (__bf16*)take((size_t)N * F * 2);       // bf16 A (layer1: x, layer2: h)
    __bf16* W1t  = (__bf16*)take((size_t)F * F * 2);       // transposed [256 x 256]
    __bf16* W2t  = (__bf16*)take((size_t)Cpad * F * 2);    // transposed+padded [48 x 256]
    float*  hlin = (float*) take((size_t)N * F * 4);       // GEMM out (layer2 stride Cpad)
    float*  h    = (float*) take((size_t)N * F * 4);       // aggregated hidden
    float*  nrm  = (float*) take((size_t)N * 4);
    float*  sim  = (float*) take((size_t)E * 4);
    float*  rsum = (float*) take((size_t)N * 4);
    float*  deg  = (float*) take((size_t)N * 4);
    float*  wsv  = (float*) take((size_t)N * 4);
    float*  we   = (float*) take((size_t)E * 4);
    float*  wdeg = (float*) take((size_t)N * 4);
    float*  dinv = (float*) take((size_t)N * 4);

    const int TB = 256;
    const long long NF = (long long)N * F;
    const int nodeBlocks  = (N + TB - 1) / TB;
    const int edgeBlocks  = (E + TB - 1) / TB;
    const int waveNodeBlk = (N + 7) / 8;   // 8 waves per 256-thread block
    const int waveEdgeBlk = (E + 7) / 8;
    const int nfBlocks    = (int)((NF + TB - 1) / TB);

    // ---- weight conversion (transposed for contiguous-K WMMA B loads) ----
    transpose_w_bf16_kernel<<<(F * F + TB - 1) / TB, TB, 0, stream>>>(W1, W1t, F, F, F);
    transpose_w_bf16_kernel<<<(Cpad * F + TB - 1) / TB, TB, 0, stream>>>(W2, W2t, F, C, Cpad);
    cvt_bf16_kernel<<<nfBlocks, TB, 0, stream>>>(x, xb, NF);

    // ================= layer 1 =================
    hipMemsetAsync(rsum, 0, (size_t)N * 4, stream);
    hipMemsetAsync(deg,  0, (size_t)N * 4, stream);

    norms_kernel<<<waveNodeBlk, TB, 0, stream>>>(x, nrm, N, F);
    edge_sim_kernel<<<waveEdgeBlk, TB, 0, stream>>>(x, row, col, nrm, sim, rsum, E, F);
    edge_norm_kernel<<<edgeBlocks, TB, 0, stream>>>(sim, row, rsum, deg, E);
    node_ws_kernel<<<nodeBlocks, TB, 0, stream>>>(deg, wsv, wdeg, N);
    edge_we_kernel<<<edgeBlocks, TB, 0, stream>>>(sim, col, we, wdeg, E);
    node_dinv_kernel<<<nodeBlocks, TB, 0, stream>>>(wdeg, dinv, N);

    {   // h_lin = x @ W1: 16 N-tiles = 4 waves x NT=4 ; one block per 16-row stripe
        dim3 blk(32, 4), grd(1, N / 16);
        gemm_bf16_wmma_kernel<256, 4><<<grd, blk, 0, stream>>>(xb, W1t, hlin, F);
    }
    agg_init_kernel<<<nfBlocks, TB, 0, stream>>>(hlin, dinv, wsv, b1, h, N, F, F, F);
    agg_edges_kernel<<<waveEdgeBlk, TB, 0, stream>>>(hlin, row, col, we, dinv, h, E, F, F, F);
    relu_kernel<<<nfBlocks, TB, 0, stream>>>(h, NF);

    // ================= layer 2 =================
    cvt_bf16_kernel<<<nfBlocks, TB, 0, stream>>>(h, xb, NF);   // reuse xb as bf16(h)
    hipMemsetAsync(rsum, 0, (size_t)N * 4, stream);
    hipMemsetAsync(deg,  0, (size_t)N * 4, stream);

    norms_kernel<<<waveNodeBlk, TB, 0, stream>>>(h, nrm, N, F);
    edge_sim_kernel<<<waveEdgeBlk, TB, 0, stream>>>(h, row, col, nrm, sim, rsum, E, F);
    edge_norm_kernel<<<edgeBlocks, TB, 0, stream>>>(sim, row, rsum, deg, E);
    node_ws_kernel<<<nodeBlocks, TB, 0, stream>>>(deg, wsv, wdeg, N);
    edge_we_kernel<<<edgeBlocks, TB, 0, stream>>>(sim, col, we, wdeg, E);
    node_dinv_kernel<<<nodeBlocks, TB, 0, stream>>>(wdeg, dinv, N);

    {   // l_lin = h @ W2 (Cpad=48 cols): one wave per stripe, NT=3 tiles
        dim3 blk(32, 1), grd(1, N / 16);
        gemm_bf16_wmma_kernel<256, 3><<<grd, blk, 0, stream>>>(xb, W2t, hlin, Cpad);
    }
    // aggregate directly into d_out [N x C]
    long long NC = (long long)N * C;
    agg_init_kernel<<<(int)((NC + TB - 1) / TB), TB, 0, stream>>>(
        hlin, dinv, wsv, b2, out, N, C, Cpad, C);
    agg_edges_kernel<<<waveEdgeBlk, TB, 0, stream>>>(
        hlin, row, col, we, dinv, out, E, C, Cpad, C);
    logsoftmax_kernel<<<nodeBlocks, TB, 0, stream>>>(out, N, C);
}
